// NeuralBellmanFordNetwork_28346784154118
// MI455X (gfx1250) — matrix operand order repaired and checked by reading, not compiled, not run on hardware
//
#include <hip/hip_runtime.h>
#include <hip/hip_bf16.h>
#include <math.h>

// ---------------- problem constants (match reference) ----------------
#define NN      50000
#define RR      100
#define DD      32
#define LL      6
#define BB      4
#define KK      33
#define FF      64          // 2*D
#define TWOR    (2*RR)
#define TRD     (TWOR*DD)   // 6400
#define THIRTEEND (13*DD)   // 416
#define NBD     (NN*BB*DD)  // 6,400,000
#define ROWS    (NN*BB)     // 200,000
#define EPSV    1e-06f

typedef __attribute__((ext_vector_type(16))) __bf16 v16bf;
typedef __attribute__((ext_vector_type(8)))  float  v8f;

union FragU { uint4 u[2]; v16bf v; };

static __device__ __forceinline__ unsigned short f2bf(float f) {
    unsigned u = __float_as_uint(f);
    u += 0x7FFFu + ((u >> 16) & 1u);   // round-to-nearest-even
    return (unsigned short)(u >> 16);
}

// sign-aware float atomic max/min via int/uint ordering
static __device__ __forceinline__ void atomicMaxF(float* a, float v) {
    if (v >= 0.f) atomicMax((int*)a, __float_as_int(v));
    else          atomicMin((unsigned int*)a, __float_as_uint(v));
}
static __device__ __forceinline__ void atomicMinF(float* a, float v) {
    if (v >= 0.f) atomicMin((int*)a, __float_as_int(v));
    else          atomicMax((unsigned int*)a, __float_as_uint(v));
}

// ---------------- tiny utility kernels ----------------
__global__ void k_fill_f32(float* p, float v, long n) {
    long i = (long)blockIdx.x * blockDim.x + threadIdx.x;
    long stride = (long)gridDim.x * blockDim.x;
    for (; i < n; i += stride) p[i] = v;
}

__global__ void k_init_acc(float* s, float* q, float* mx, float* mn) {
    int i = blockIdx.x * blockDim.x + threadIdx.x;  // exact grid: NBD
    s[i] = 0.f; q[i] = 0.f; mx[i] = -INFINITY; mn[i] = INFINITY;
}

// setup: is_t_neg flags, h0, r0->query gather, t-selection
__global__ void k_setup(const int* __restrict__ hI, const int* __restrict__ tI,
                        const int* __restrict__ rI, const float* __restrict__ qemb,
                        int* h0, int* tsel, float* query) {
    __shared__ int flag_s[BB], h0_s[BB], r0_s[BB];
    int tid = threadIdx.x;
    if (tid < BB) {
        int base = tid * KK, h00 = hI[base], f = 1;
        for (int k = 1; k < KK; ++k) if (hI[base + k] != h00) f = 0;
        flag_s[tid] = f;
        h0_s[tid] = f ? h00 : tI[base];
        r0_s[tid] = f ? rI[base] : rI[base] + RR;
        h0[tid] = h0_s[tid];
    }
    __syncthreads();
    if (tid < BB * KK) {
        int b = tid / KK, k = tid % KK;
        tsel[tid] = flag_s[b] ? tI[b * KK + k] : hI[b * KK + k];
    }
    if (tid < BB * DD) {
        int b = tid >> 5, d = tid & 31;
        query[tid] = qemb[r0_s[b] * DD + d];
    }
}

// hidden(layer0) = boundary: scatter query rows into zeroed buffer
__global__ void k_scatter_boundary(const int* __restrict__ h0,
                                   const float* __restrict__ query, float* hid) {
    int tid = threadIdx.x;           // 128 threads
    int b = tid >> 5, d = tid & 31;
    hid[((long)h0[b] * BB + b) * DD + d] = query[tid];
}

// cnt[dst] += 1 per edge (cnt pre-filled with 1.0 self-loop)
__global__ void k_degree(const int* __restrict__ edges, float* cnt, int E) {
    int e = blockIdx.x * blockDim.x + threadIdx.x;
    if (e < E) atomicAdd(&cnt[edges[3 * e + 1]], 1.0f);
}

// deterministic two-pass mean of log(cnt)
__global__ void k_logsum_partial(const float* __restrict__ cnt, float* partials) {
    __shared__ float red[256];
    float s = 0.f;
    for (int i = blockIdx.x * blockDim.x + threadIdx.x; i < NN; i += gridDim.x * blockDim.x)
        s += logf(cnt[i]);
    red[threadIdx.x] = s; __syncthreads();
    for (int off = 128; off > 0; off >>= 1) {
        if (threadIdx.x < off) red[threadIdx.x] += red[threadIdx.x + off];
        __syncthreads();
    }
    if (threadIdx.x == 0) partials[blockIdx.x] = red[0];
}
__global__ void k_logsum_final(const float* __restrict__ partials, float* meanv) {
    __shared__ float red[256];
    red[threadIdx.x] = partials[threadIdx.x]; __syncthreads();
    for (int off = 128; off > 0; off >>= 1) {
        if (threadIdx.x < off) red[threadIdx.x] += red[threadIdx.x + off];
        __syncthreads();
    }
    if (threadIdx.x == 0) meanv[0] = red[0] / (float)NN;
}
__global__ void k_scales(const float* __restrict__ cnt, const float* __restrict__ meanv,
                         float* scaleS, float* scaleI) {
    int n = blockIdx.x * blockDim.x + threadIdx.x;
    if (n < NN) {
        float s = logf(cnt[n]) / meanv[0];
        scaleS[n] = s;
        scaleI[n] = 1.0f / fmaxf(s, 0.01f);
    }
}

// rel_emb[l] = (query @ rel_W[l] + rel_b[l]) laid out (2R, B, D)
__global__ void k_relemb(int l, const float* __restrict__ query,
                         const float* __restrict__ relW, const float* __restrict__ relb,
                         float* relemb) {
    int o = blockIdx.x * blockDim.x + threadIdx.x;     // exact: 2R*B*D = 25600
    int rel = o / (BB * DD);
    int rem = o - rel * (BB * DD);
    int b = rem >> 5, d = rem & 31;
    int j = rel * DD + d;
    float s = relb[l * TRD + j];
    const float* w = relW + (long)l * DD * TRD + j;
    const float* q = query + b * DD;
#pragma unroll
    for (int i = 0; i < DD; ++i) s += q[i] * w[(long)i * TRD];
    relemb[o] = s;
}

// edge message + segment reductions: one wave per edge, lane = d
__global__ void k_edges(const int* __restrict__ edges, const float* __restrict__ hidden,
                        const float* __restrict__ relemb,
                        float* accS, float* accQ, float* accMx, float* accMn, int E) {
    int gtid = blockIdx.x * blockDim.x + threadIdx.x;
    int e = gtid >> 5;
    if (e >= E) return;
    int lane = threadIdx.x & 31;
    int src = edges[3 * e], dst = edges[3 * e + 1], rel = edges[3 * e + 2];
    const float* h = hidden + ((long)src * BB) * DD + lane;
    const float* r = relemb + ((long)rel * BB) * DD + lane;
    long obase = ((long)dst * BB) * DD + lane;
#pragma unroll
    for (int b = 0; b < BB; ++b) {
        float m = h[b * DD] * r[b * DD];
        long o = obase + b * DD;
        atomicAdd(&accS[o], m);
        atomicAdd(&accQ[o], m * m);
        atomicMaxF(&accMx[o], m);
        atomicMinF(&accMn[o], m);
    }
}

// pre-pack lin_W (all layers) into per-lane bf16 B fragments
// idx = ((l*13+kk)*2+t)*32+lane ; lane holds column n=t*16+(lane&15),
// 16 consecutive K starting at kk*32 + (lane<16 ? 0 : 16)
__global__ void k_packW(const float* __restrict__ linW, unsigned short* Wpack) {
    int idx = blockIdx.x * blockDim.x + threadIdx.x;
    if (idx >= LL * 13 * 2 * 32) return;
    int lane = idx & 31;
    int t = (idx >> 5) & 1;
    int kk = (idx >> 6) % 13;
    int l = idx / (13 * 2 * 32);
    int n = t * 16 + (lane & 15);
    int kb = kk * 32 + ((lane & 16) ? 16 : 0);
    unsigned short* dst = Wpack + (long)idx * 16;
#pragma unroll
    for (int j = 0; j < 16; ++j)
        dst[j] = f2bf(linW[((long)l * THIRTEEND + kb + j) * DD + n]);
}

// ---------------- fused stats -> features -> bf16 WMMA GEMM -> ReLU ----------------
// block: 256 threads (8 waves), 4 M-tiles of 16 rows (rows = node*B+b), N=32 cols
#define ALDS_STRIDE 424   // 416 + 8 pad: row-to-row offset 212 dwords, conflict-free
__global__ void k_update(int l,
                         const float* __restrict__ hidden,
                         const float* __restrict__ accS, const float* __restrict__ accQ,
                         const float* __restrict__ accMx, const float* __restrict__ accMn,
                         const float* __restrict__ cnt,
                         const float* __restrict__ scaleS, const float* __restrict__ scaleI,
                         const int* __restrict__ h0, const float* __restrict__ query,
                         const unsigned short* __restrict__ Wpack,
                         const float* __restrict__ linb,
                         float* __restrict__ out) {
    __shared__ unsigned short Alds[4 * 16 * ALDS_STRIDE];   // 54272 B
    int tid = threadIdx.x;
    long rowbase = (long)blockIdx.x * 64;

    // ----- phase 1: build bf16 A rows [hidden | (feats x scales)] -----
#pragma unroll
    for (int it = 0; it < 8; ++it) {
        int p = it * 256 + tid;            // 2048 (row_local, d) pairs
        int row_local = p >> 5, d = p & 31;
        long g = rowbase + row_local;
        int n = (int)(g >> 2), b = (int)(g & 3);
        long o = g * DD + d;
        float hv = hidden[o];
        float bnd = (n == h0[b]) ? query[b * DD + d] : 0.f;
        float c  = cnt[n];
        float mean = (accS[o] + bnd) / c;
        float sqm  = (accQ[o] + bnd * bnd) / c;
        float mxv  = fmaxf(accMx[o], bnd);
        float mnv  = fminf(accMn[o], bnd);
        float sd   = sqrtf(fmaxf(sqm - mean * mean, EPSV));
        float s1 = scaleS[n], s2 = scaleI[n];
        unsigned short* arow = Alds + (long)row_local * ALDS_STRIDE;
        arow[d] = f2bf(hv);
        unsigned short* u = arow + DD + d * 12;
        float f4[4] = {mean, mxv, mnv, sd};
#pragma unroll
        for (int s = 0; s < 4; ++s) {
            u[s * 3 + 0] = f2bf(f4[s]);
            u[s * 3 + 1] = f2bf(f4[s] * s1);
            u[s * 3 + 2] = f2bf(f4[s] * s2);
        }
    }
    __syncthreads();

    // ----- phase 2: 2 waves per tile, 13 x v_wmma_f32_16x16x32_bf16 each -----
    int wave = tid >> 5, lane = tid & 31;
    int tile = wave >> 1, t = wave & 1;
    int row16 = lane & 15;
    int half = (lane & 16) ? 16 : 0;
    const unsigned short* arow = Alds + ((long)tile * 16 + row16) * ALDS_STRIDE;
    const unsigned short* bbase = Wpack + (((long)(l * 13) * 2 + t) * 32 + lane) * 16;

    v8f acc = {};
#pragma unroll
    for (int kk = 0; kk < 13; ++kk) {
        FragU a, b;
        a.u[0] = *(const uint4*)(arow + kk * 32 + half);
        a.u[1] = *(const uint4*)(arow + kk * 32 + half + 8);
        const unsigned short* bp = bbase + (long)kk * 2 * 32 * 16;
        b.u[0] = *(const uint4*)(bp);
        b.u[1] = *(const uint4*)(bp + 8);
        acc = __builtin_amdgcn_wmma_f32_16x16x32_bf16(false, a.v, false, b.v,
                                                      (short)0, acc, false, false);
    }

    int col = t * 16 + row16;
    float bias = linb[l * DD + col];
    int mofs = (lane & 16) ? 8 : 0;
#pragma unroll
    for (int r = 0; r < 8; ++r) {
        long g = rowbase + tile * 16 + r + mofs;
        out[g * DD + col] = fmaxf(acc[r] + bias, 0.f);
    }
}

// ---------------- final readout MLP: one block per (b,k) ----------------
__global__ void k_score(const float* __restrict__ hidden, const float* __restrict__ query,
                        const int* __restrict__ tsel,
                        const float* __restrict__ W1, const float* __restrict__ b1,
                        const float* __restrict__ W2, const float* __restrict__ b2,
                        float* __restrict__ outp) {
    __shared__ float feat[FF];
    __shared__ float red[FF];
    int bk = blockIdx.x;
    int b = bk / KK;
    int j = threadIdx.x;    // 64 threads
    int tnode = tsel[bk];
    feat[j] = (j < DD) ? hidden[((long)tnode * BB + b) * DD + j]
                       : query[b * DD + (j - DD)];
    __syncthreads();
    float h = b1[j];
#pragma unroll
    for (int i = 0; i < FF; ++i) h += feat[i] * W1[i * FF + j];
    h = fmaxf(h, 0.f);
    red[j] = h * W2[j];
    __syncthreads();
    for (int off = 32; off > 0; off >>= 1) {
        if (j < off) red[j] += red[j + off];
        __syncthreads();
    }
    if (j == 0) outp[bk] = red[0] + b2[0];
}

// ---------------- host-side launch ----------------
extern "C" void kernel_launch(void* const* d_in, const int* in_sizes, int n_in,
                              void* d_out, int out_size, void* d_ws, size_t ws_size,
                              hipStream_t stream) {
    const int*   edges = (const int*)d_in[0];
    const int*   hIdx  = (const int*)d_in[1];
    const int*   tIdx  = (const int*)d_in[2];
    const int*   rIdx  = (const int*)d_in[3];
    const float* qemb  = (const float*)d_in[4];
    const float* relW  = (const float*)d_in[5];
    const float* relb  = (const float*)d_in[6];
    const float* linW  = (const float*)d_in[7];
    const float* linb  = (const float*)d_in[8];
    const float* W1    = (const float*)d_in[9];
    const float* b1v   = (const float*)d_in[10];
    const float* W2    = (const float*)d_in[11];
    const float* b2v   = (const float*)d_in[12];
    const int E = in_sizes[0] / 3;   // 500000

    // workspace sub-allocation (~155 MB total; L2-resident on MI455X)
    char* ws = (char*)d_ws;
    size_t off = 0;
    auto alloc = [&](size_t bytes) {
        void* p = ws + off;
        off = (off + bytes + 255) & ~(size_t)255;
        return p;
    };
    float* hiddenA = (float*)alloc((size_t)NBD * 4);
    float* hiddenB = (float*)alloc((size_t)NBD * 4);
    float* accS    = (float*)alloc((size_t)NBD * 4);
    float* accQ    = (float*)alloc((size_t)NBD * 4);
    float* accMx   = (float*)alloc((size_t)NBD * 4);
    float* accMn   = (float*)alloc((size_t)NBD * 4);
    float* relemb  = (float*)alloc((size_t)TWOR * BB * DD * 4);
    float* cnt     = (float*)alloc((size_t)NN * 4);
    float* scaleS  = (float*)alloc((size_t)NN * 4);
    float* scaleI  = (float*)alloc((size_t)NN * 4);
    float* query   = (float*)alloc(BB * DD * 4);
    int*   h0      = (int*)alloc(BB * 4);
    int*   tsel    = (int*)alloc(BB * KK * 4);
    float* partials= (float*)alloc(256 * 4);
    float* meanv   = (float*)alloc(4);
    unsigned short* Wpack = (unsigned short*)alloc((size_t)LL * 13 * 2 * 32 * 16 * 2);
    (void)ws_size; (void)n_in; (void)out_size;

    // setup: flags / h0 / query / t-selection
    k_setup<<<1, 256, 0, stream>>>(hIdx, tIdx, rIdx, qemb, h0, tsel, query);

    // hidden(0) = boundary
    k_fill_f32<<<4096, 256, 0, stream>>>(hiddenA, 0.f, (long)NBD);
    k_scatter_boundary<<<1, 128, 0, stream>>>(h0, query, hiddenA);

    // degree / scales (deterministic reduction)
    k_fill_f32<<<256, 256, 0, stream>>>(cnt, 1.0f, (long)NN);
    k_degree<<<(E + 255) / 256, 256, 0, stream>>>(edges, cnt, E);
    k_logsum_partial<<<256, 256, 0, stream>>>(cnt, partials);
    k_logsum_final<<<1, 256, 0, stream>>>(partials, meanv);
    k_scales<<<(NN + 255) / 256, 256, 0, stream>>>(cnt, meanv, scaleS, scaleI);

    // pre-pack weights into WMMA B-fragment layout (bf16)
    k_packW<<<(LL * 13 * 2 * 32 + 255) / 256, 256, 0, stream>>>(linW, Wpack);

    float* hin = hiddenA;
    float* hout = hiddenB;
    for (int l = 0; l < LL; ++l) {
        k_relemb<<<TWOR * BB * DD / 256, 256, 0, stream>>>(l, query, relW, relb, relemb);
        k_init_acc<<<NBD / 256, 256, 0, stream>>>(accS, accQ, accMx, accMn);
        k_edges<<<(E * 32 + 255) / 256, 256, 0, stream>>>(edges, hin, relemb,
                                                          accS, accQ, accMx, accMn, E);
        k_update<<<ROWS / 64, 256, 0, stream>>>(l, hin, accS, accQ, accMx, accMn,
                                                cnt, scaleS, scaleI, h0, query,
                                                Wpack, linb, hout);
        float* tmp = hin; hin = hout; hout = tmp;
    }

    // final scoring MLP
    k_score<<<BB * KK, FF, 0, stream>>>(hin, query, tsel, W1, b1v, W2, b2v, (float*)d_out);
}
